// MultiHeadAttention_17154099381033
// MI455X (gfx1250) — matrix operand order repaired and checked by reading, not compiled, not run on hardware
//
#include <hip/hip_runtime.h>

// ---------------------------------------------------------------------------
// MHA forward for MI455X (gfx1250, wave32).
//  - all matmuls via v_wmma_f32_16x16x32_bf16 (fp32 accumulate)
//  - projection weight tiles staged by the Tensor Data Mover (TENSORcnt)
//  - attention K/V tiles double-buffered via global_load_async_to_lds_b128
//    (ASYNCcnt), FA2-style block cooperation, fp32 online softmax
// ---------------------------------------------------------------------------

typedef __bf16 bf16_t;
typedef __attribute__((ext_vector_type(16))) __bf16 v16bf;
typedef __attribute__((ext_vector_type(8)))  __bf16 v8bf;
typedef __attribute__((ext_vector_type(8)))  float  v8f;
typedef __attribute__((ext_vector_type(4)))  unsigned su4;
typedef __attribute__((ext_vector_type(8)))  unsigned su8;

#define DEV_INLINE __device__ __forceinline__

constexpr int Bn = 2, Hn = 8, Sn = 4096, Dn = 64, Fn = 512;
constexpr int Mtot = Bn * Sn;                              // 8192 rows
constexpr size_t HeadElems = (size_t)Bn * Hn * Sn * Dn;    // 4M elems
constexpr int WROW = 65;   // padded LDS row for TDM fp32 weight tile (DWORDs)
constexpr int KROW = 72;   // padded LDS row for K tile (bf16; 144B, 16B-aligned)
constexpr int VROW = 40;   // padded LDS row for V tile (bf16; 80B, 16B-aligned)

DEV_INLINE v8f wmma_bf16(v16bf a, v16bf b, v8f c) {
  return __builtin_amdgcn_wmma_f32_16x16x32_bf16(false, a, false, b, (short)0, c,
                                                 false, false);
}

// ---------------------------------------------------------------------------
// TDM: DMA a 32(K) x 64(N) fp32 tile of a row-major [*,512] matrix into LDS.
// pad_enable inserts 1 DWORD after every 64 DWORDs -> LDS row stride = 65.
// ---------------------------------------------------------------------------
DEV_INLINE void tdm_load_w_tile(const float* gsrc, float* lds_dst) {
  unsigned long long ga = (unsigned long long)gsrc;
  unsigned ga_lo = __builtin_amdgcn_readfirstlane((unsigned)ga);
  unsigned ga_hi = __builtin_amdgcn_readfirstlane((unsigned)(ga >> 32));
  unsigned lds_a = __builtin_amdgcn_readfirstlane(
      (unsigned)(unsigned long long)(void*)lds_dst);

  su4 g0;
  g0[0] = 1u;                                  // count=1
  g0[1] = lds_a;                               // lds_addr
  g0[2] = ga_lo;                               // global_addr lo
  g0[3] = (ga_hi & 0x01FFFFFFu) | 0x80000000u; // global_addr[56:32] | type=2

  su8 g1;
  g1[0] = (2u << 16) | (1u << 20) | (5u << 22); // 4B elems, pad 1 DW / 64 DW
  g1[1] = 0xFFFF0000u;     // tensor_dim0 = 0xFFFF (tile interior)
  g1[2] = 0xFFFF0000u;     // tensor_dim1 = 0xFFFF
  g1[3] = (64u << 16);     // tile_dim0 = 64
  g1[4] = 32u;             // tile_dim1 = 32
  g1[5] = (unsigned)Fn;    // tensor_dim0_stride = 512
  g1[6] = 0u;
  g1[7] = 0u;

  asm volatile("tensor_load_to_lds %0, %1" : : "s"(g0), "s"(g1) : "memory");
}

// Async global -> LDS, 16 bytes per lane, ASYNCcnt-tracked.
DEV_INLINE void async_ld_b128(const void* gsrc, void* lds_dst) {
  unsigned lds_off = (unsigned)(unsigned long long)lds_dst;
  asm volatile("global_load_async_to_lds_b128 %0, %1, off"
               : : "v"(lds_off), "v"(gsrc) : "memory");
}

// A fragment (16x32, MxK) from fp32 row-major [*, ld] at (row0, k0).
DEV_INLINE v16bf load_a_f32(const float* __restrict__ base, int row0, int ld,
                            int k0, int r, int hi) {
  const float* p = base + (size_t)(row0 + r) * ld + k0 + hi * 8;
  v16bf a;
#pragma unroll
  for (int i = 0; i < 8; ++i) a[i] = (__bf16)p[i];
#pragma unroll
  for (int i = 0; i < 8; ++i) a[8 + i] = (__bf16)p[16 + i];
  return a;
}

// A fragment from a bf16 row-major matrix (two b128 loads per lane).
DEV_INLINE v16bf load_a_bf16(const bf16_t* __restrict__ base, int row0, int ld,
                             int k0, int r, int hi) {
  const bf16_t* p = base + (size_t)(row0 + r) * ld + k0 + hi * 8;
  v8bf lo = *(const v8bf*)p;
  v8bf hh = *(const v8bf*)(p + 16);
  return __builtin_shufflevector(lo, hh, 0, 1, 2, 3, 4, 5, 6, 7, 8, 9, 10, 11,
                                 12, 13, 14, 15);
}

// B fragment (32x16, KxN), contraction dim contiguous: elem (k,n) at
// base[(col0+n)*ld + k0 + k].  Works for global bf16 and LDS-staged tiles.
DEV_INLINE v16bf load_b_kcontig_bf16(const bf16_t* base, int col0, int ld,
                                     int k0, int r, int hi) {
  const bf16_t* p = base + (size_t)(col0 + r) * ld + k0 + hi * 8;
  v8bf lo = *(const v8bf*)p;
  v8bf hh = *(const v8bf*)(p + 16);
  return __builtin_shufflevector(lo, hh, 0, 1, 2, 3, 4, 5, 6, 7, 8, 9, 10, 11,
                                 12, 13, 14, 15);
}

// B fragment from the TDM-staged LDS fp32 tile [32][WROW].
DEV_INLINE v16bf load_b_lds_f32(const float* lds_w, int colLocal, int r, int hi) {
  const float* p = lds_w + hi * 8 * WROW + colLocal + r;
  v16bf b;
#pragma unroll
  for (int i = 0; i < 16; ++i) b[i] = (__bf16)p[(i + (i >> 3) * 8) * WROW];
  return b;
}

// ---------------------------------------------------------------------------
// Kernel 1: fused Q/K/V projection (TDM-staged weight tiles).
// ---------------------------------------------------------------------------
__global__ __launch_bounds__(256) void qkv_proj_kernel(
    const float* __restrict__ Xq, const float* __restrict__ Xk,
    const float* __restrict__ Xv, const float* __restrict__ Wq,
    const float* __restrict__ Wk, const float* __restrict__ Wv,
    const float* __restrict__ Bq, const float* __restrict__ Bk,
    const float* __restrict__ Bv, bf16_t* __restrict__ Qh,
    bf16_t* __restrict__ Kh, bf16_t* __restrict__ Vt) {
  __shared__ float lds_w[32 * WROW];

  const int z = blockIdx.z;
  const float* X  = (z == 0) ? Xq : (z == 1) ? Xk : Xv;
  const float* W  = (z == 0) ? Wq : (z == 1) ? Wk : Wv;
  const float* Bi = (z == 0) ? Bq : (z == 1) ? Bk : Bv;

  const int lane = threadIdx.x & 31;
  const int wave = threadIdx.x >> 5;
  const int r = lane & 15, hi = lane >> 4;
  const int m0 = blockIdx.x * 128 + (wave >> 1) * 32;
  const int n0b = blockIdx.y * 64;
  const int nloc = (wave & 1) * 32;

  v8f acc[2][2] = {};

  for (int kk = 0; kk < Fn; kk += 32) {
    if (wave == 0) {
      tdm_load_w_tile(W + (size_t)kk * Fn + n0b, lds_w);
      __builtin_amdgcn_s_wait_tensorcnt(0);
    }
    __syncthreads();

    v16bf af[2];
#pragma unroll
    for (int mt = 0; mt < 2; ++mt)
      af[mt] = load_a_f32(X, m0 + mt * 16, Fn, kk, r, hi);

    v16bf bfm[2];
#pragma unroll
    for (int nt = 0; nt < 2; ++nt)
      bfm[nt] = load_b_lds_f32(lds_w, nloc + nt * 16, r, hi);

#pragma unroll
    for (int mt = 0; mt < 2; ++mt)
#pragma unroll
      for (int nt = 0; nt < 2; ++nt)
        acc[mt][nt] = wmma_bf16(af[mt], bfm[nt], acc[mt][nt]);

    if (kk + 32 < Fn)
      __builtin_prefetch(X + (size_t)(m0 + r) * Fn + kk + 32, 0, 1);
    __syncthreads();
  }

#pragma unroll
  for (int mt = 0; mt < 2; ++mt)
#pragma unroll
    for (int nt = 0; nt < 2; ++nt)
#pragma unroll
      for (int j = 0; j < 8; ++j) {
        const int row = m0 + mt * 16 + hi * 8 + j;
        const int col = n0b + nloc + nt * 16 + r;
        const float v = acc[mt][nt][j] + Bi[col];
        const int b = row >> 12, s = row & (Sn - 1);
        const int h = col >> 6, d = col & (Dn - 1);
        if (z == 0)
          Qh[(((size_t)b * Hn + h) * Sn + s) * Dn + d] = (__bf16)v;
        else if (z == 1)
          Kh[(((size_t)b * Hn + h) * Sn + s) * Dn + d] = (__bf16)v;
        else
          Vt[(((size_t)b * Hn + h) * Dn + d) * Sn + s] = (__bf16)v;
      }
}

// ---------------------------------------------------------------------------
// Kernel 2: causal flash attention, block-cooperative.
// Block = 8 waves = 128 queries of one (b,h).  K/V tiles (32 keys) are
// double-buffered in LDS via async global->LDS loads issued one chunk ahead.
// ---------------------------------------------------------------------------
DEV_INLINE float half_reduce_max(float v) {
#pragma unroll
  for (int off = 1; off < 16; off <<= 1) v = fmaxf(v, __shfl_xor(v, off, 32));
  return v;
}
DEV_INLINE float half_reduce_sum(float v) {
#pragma unroll
  for (int off = 1; off < 16; off <<= 1) v += __shfl_xor(v, off, 32);
  return v;
}

__global__ __launch_bounds__(256) void attn_kernel(
    const bf16_t* __restrict__ Qh, const bf16_t* __restrict__ Kh,
    const bf16_t* __restrict__ Vt, bf16_t* __restrict__ Att) {
  __shared__ bf16_t ldsK[2][32 * KROW];   // [key][d], padded rows
  __shared__ bf16_t ldsV[2][64 * VROW];   // [d][key], padded rows
  __shared__ bf16_t lds_p[8 * 16 * 32];   // per-wave P staging

  const int t = threadIdx.x;
  const int lane = t & 31;
  const int wave = t >> 5;
  const int r = lane & 15, hi = lane >> 4;

  const int bh = blockIdx.x >> 5;           // 32 query-blocks per (b,h)
  const int q0b = (blockIdx.x & 31) * 128;  // block's first query
  const int q0 = q0b + wave * 16;           // wave's first query
  const int b = bh >> 3, h = bh & 7;

  const bf16_t* Qb = Qh + (size_t)bh * Sn * Dn;
  const bf16_t* Kb = Kh + (size_t)bh * Sn * Dn;
  const bf16_t* Vb = Vt + (size_t)bh * Dn * Sn;
  bf16_t* lp = lds_p + wave * 16 * 32;

  // cooperative stage of one 32-key chunk into buffer `buf`
  const int krow = t >> 3, kseg = t & 7;    // K: 32 rows x 128B
  const int vrow = t >> 2, vseg = t & 3;    // V: 64 rows x 64B
  auto stage = [&](int buf, int c0) {
    async_ld_b128(Kb + (size_t)(c0 + krow) * Dn + kseg * 8,
                  &ldsK[buf][krow * KROW + kseg * 8]);
    async_ld_b128(Vb + (size_t)vrow * Sn + c0 + vseg * 8,
                  &ldsV[buf][vrow * VROW + vseg * 8]);
  };

  v16bf qf[2];
#pragma unroll
  for (int tt = 0; tt < 2; ++tt) qf[tt] = load_a_bf16(Qb, q0, Dn, tt * 32, r, hi);

  v8f acc[4] = {};
  float mrow[8], lrow[8];
#pragma unroll
  for (int j = 0; j < 8; ++j) { mrow[j] = -1e30f; lrow[j] = 0.0f; }

  const float scale = 0.125f;  // 1/sqrt(64)
  const int cEnd = q0b + 128;  // block causal frontier

  stage(0, 0);
  int cur = 0;
  for (int c0 = 0; c0 < cEnd; c0 += 32, cur ^= 1) {
    asm volatile("s_wait_asynccnt 0x0" ::: "memory");
    __syncthreads();                       // tile ready; prev readers done
    if (c0 + 32 < cEnd) stage(cur ^ 1, c0 + 32);

    if (c0 < q0 + 16) {                    // wave-uniform: has unmasked keys
      v8f s0 = {}, s1 = {};
#pragma unroll
      for (int ks = 0; ks < 2; ++ks) {
        v16bf kf0 = load_b_kcontig_bf16(ldsK[cur], 0, KROW, ks * 32, r, hi);
        v16bf kf1 = load_b_kcontig_bf16(ldsK[cur], 16, KROW, ks * 32, r, hi);
        s0 = wmma_bf16(qf[ks], kf0, s0);
        s1 = wmma_bf16(qf[ks], kf1, s1);
      }

      float p0[8], p1[8];
#pragma unroll
      for (int j = 0; j < 8; ++j) {
        const int qrow = q0 + hi * 8 + j;
        p0[j] = (c0 + r <= qrow) ? s0[j] * scale : -1e30f;
        p1[j] = (c0 + 16 + r <= qrow) ? s1[j] * scale : -1e30f;
      }
#pragma unroll
      for (int j = 0; j < 8; ++j) {
        const float mx = half_reduce_max(fmaxf(p0[j], p1[j]));
        const float mnew = fmaxf(mrow[j], mx);
        const float corr = __expf(mrow[j] - mnew);
        mrow[j] = mnew;
        p0[j] = __expf(p0[j] - mnew);
        p1[j] = __expf(p1[j] - mnew);
        const float rs = half_reduce_sum(p0[j] + p1[j]);
        lrow[j] = lrow[j] * corr + rs;
        acc[0][j] *= corr; acc[1][j] *= corr;
        acc[2][j] *= corr; acc[3][j] *= corr;
      }

      // C-layout P -> A-fragment via wave-private LDS
#pragma unroll
      for (int j = 0; j < 8; ++j) {
        const int prow = hi * 8 + j;
        lp[prow * 32 + r]      = (__bf16)p0[j];
        lp[prow * 32 + 16 + r] = (__bf16)p1[j];
      }
      asm volatile("s_wait_dscnt 0x0" ::: "memory");
      v16bf pa;
#pragma unroll
      for (int i = 0; i < 8; ++i) pa[i] = lp[r * 32 + hi * 8 + i];
#pragma unroll
      for (int i = 0; i < 8; ++i) pa[8 + i] = lp[r * 32 + 16 + hi * 8 + i];

#pragma unroll
      for (int dt = 0; dt < 4; ++dt) {
        v16bf vf = load_b_kcontig_bf16(ldsV[cur], dt * 16, VROW, 0, r, hi);
        acc[dt] = wmma_bf16(pa, vf, acc[dt]);
      }
    }
  }

  float inv[8];
#pragma unroll
  for (int j = 0; j < 8; ++j) inv[j] = 1.0f / lrow[j];
#pragma unroll
  for (int dt = 0; dt < 4; ++dt)
#pragma unroll
    for (int j = 0; j < 8; ++j) {
      const int s = q0 + hi * 8 + j;
      const int d = dt * 16 + r;
      Att[((size_t)b * Sn + s) * Fn + h * Dn + d] = (__bf16)(acc[dt][j] * inv[j]);
    }
}

// ---------------------------------------------------------------------------
// Kernel 3: output projection (TDM-staged weight tiles), fp32 result.
// ---------------------------------------------------------------------------
__global__ __launch_bounds__(256) void out_proj_kernel(
    const bf16_t* __restrict__ A, const float* __restrict__ W,
    const float* __restrict__ Bi, float* __restrict__ out) {
  __shared__ float lds_w[32 * WROW];

  const int lane = threadIdx.x & 31;
  const int wave = threadIdx.x >> 5;
  const int r = lane & 15, hi = lane >> 4;
  const int m0 = blockIdx.x * 128 + (wave >> 1) * 32;
  const int n0b = blockIdx.y * 64;
  const int nloc = (wave & 1) * 32;

  v8f acc[2][2] = {};

  for (int kk = 0; kk < Fn; kk += 32) {
    if (wave == 0) {
      tdm_load_w_tile(W + (size_t)kk * Fn + n0b, lds_w);
      __builtin_amdgcn_s_wait_tensorcnt(0);
    }
    __syncthreads();

    v16bf af[2];
#pragma unroll
    for (int mt = 0; mt < 2; ++mt)
      af[mt] = load_a_bf16(A, m0 + mt * 16, Fn, kk, r, hi);

    v16bf bfm[2];
#pragma unroll
    for (int nt = 0; nt < 2; ++nt)
      bfm[nt] = load_b_lds_f32(lds_w, nloc + nt * 16, r, hi);

#pragma unroll
    for (int mt = 0; mt < 2; ++mt)
#pragma unroll
      for (int nt = 0; nt < 2; ++nt)
        acc[mt][nt] = wmma_bf16(af[mt], bfm[nt], acc[mt][nt]);

    if (kk + 32 < Fn)
      __builtin_prefetch(A + (size_t)(m0 + r) * Fn + kk + 32, 0, 1);
    __syncthreads();
  }

#pragma unroll
  for (int mt = 0; mt < 2; ++mt)
#pragma unroll
    for (int nt = 0; nt < 2; ++nt)
#pragma unroll
      for (int j = 0; j < 8; ++j) {
        const int row = m0 + mt * 16 + hi * 8 + j;
        const int col = n0b + nloc + nt * 16 + r;
        out[(size_t)row * Fn + col] = acc[mt][nt][j] + Bi[col];
      }
}

// ---------------------------------------------------------------------------
extern "C" void kernel_launch(void* const* d_in, const int* in_sizes, int n_in,
                              void* d_out, int out_size, void* d_ws,
                              size_t ws_size, hipStream_t stream) {
  const float* q  = (const float*)d_in[0];
  const float* k  = (const float*)d_in[1];
  const float* v  = (const float*)d_in[2];
  // d_in[3] = causal mask; structure computed analytically.
  const float* wq = (const float*)d_in[4];
  const float* bq = (const float*)d_in[5];
  const float* wk = (const float*)d_in[6];
  const float* bk = (const float*)d_in[7];
  const float* wv = (const float*)d_in[8];
  const float* bv = (const float*)d_in[9];
  const float* wo = (const float*)d_in[10];
  const float* bo = (const float*)d_in[11];
  float* out = (float*)d_out;

  bf16_t* Qh  = (bf16_t*)d_ws;
  bf16_t* Kh  = Qh + HeadElems;
  bf16_t* Vt  = Kh + HeadElems;
  bf16_t* Att = Vt + HeadElems;

  dim3 blk(256);
  dim3 gA(Mtot / 128, Fn / 64, 3);
  qkv_proj_kernel<<<gA, blk, 0, stream>>>(q, k, v, wq, wk, wv, bq, bk, bv,
                                          Qh, Kh, Vt);

  const int nQblocks = Bn * Hn * (Sn / 128);   // 512 blocks
  attn_kernel<<<dim3(nQblocks), blk, 0, stream>>>(Qh, Kh, Vt, Att);

  dim3 gC(Mtot / 128, Fn / 64);
  out_proj_kernel<<<gC, blk, 0, stream>>>(Att, wo, bo, out);
}
